// BasicSelfAttention_6468220748706
// MI455X (gfx1250) — matrix operand order repaired and checked by reading, not compiled, not run on hardware
//
#include <hip/hip_runtime.h>

typedef __attribute__((ext_vector_type(16))) _Float16 v16h;
typedef __attribute__((ext_vector_type(8)))  _Float16 v8h;
typedef __attribute__((ext_vector_type(4)))  _Float16 v4h;
typedef __attribute__((ext_vector_type(2)))  _Float16 v2h;
typedef __attribute__((ext_vector_type(8)))  float    v8f;

union U16h { v16h v; v8h h[2]; };

#define WMMA_F32_F16(A, B, C) \
  __builtin_amdgcn_wmma_f32_16x16x32_f16(false, (A), false, (B), (short)0, (C), false, false)

// ---------------------------------------------------------------------------
// f32 -> f16 conversion (vectorized x4; all sizes are multiples of 4)
// ---------------------------------------------------------------------------
__global__ __launch_bounds__(256) void cvt_f16_kernel(const float* __restrict__ in,
                                                      _Float16* __restrict__ out, int n4) {
  int i = blockIdx.x * 256 + threadIdx.x;
  if (i < n4) {
    float4 f = ((const float4*)in)[i];
    v4h hv = {(_Float16)f.x, (_Float16)f.y, (_Float16)f.z, (_Float16)f.w};
    *(v4h*)(out + (size_t)i * 4) = hv;
  }
}

// ---------------------------------------------------------------------------
// WMMA GEMM: Y[M,N] = A[M,K](f16) * W[N,K](f16)^T  (+ bias), fp32 out.
// Wave tile 32x32 (2x2 accumulators), block = 8 waves (2x4) -> 64x128 tile.
// ---------------------------------------------------------------------------
__global__ __launch_bounds__(256) void wmma_gemm_f16(const _Float16* __restrict__ A,
                                                     const _Float16* __restrict__ W,
                                                     float* __restrict__ Y,
                                                     const float* __restrict__ bias,
                                                     int M, int N, int K) {
  const int lane = threadIdx.x & 31;
  const int wave = threadIdx.x >> 5;
  const int wm = wave >> 2, wn = wave & 3;
  const int m0 = blockIdx.x * 64 + wm * 32;
  const int n0 = blockIdx.y * 128 + wn * 32;
  const int r  = lane & 15;
  const int hi = lane >> 4;      // which half of the wave
  const int kbA = hi * 8;        // A-frag K base within 16 (ISA 16-bit A layout)
  const int kbB = hi * 8;        // B-frag K base

  v8f acc[2][2] = {};
  for (int k0 = 0; k0 < K; k0 += 32) {
    U16h a[2], b[2];
#pragma unroll
    for (int i = 0; i < 2; ++i) {
      const _Float16* ap = A + (size_t)(m0 + 16 * i + r) * K + k0 + kbA;
      a[i].h[0] = *(const v8h*)(ap);        // K = kbA .. kbA+7
      a[i].h[1] = *(const v8h*)(ap + 16);   // K = 16+kbA .. 16+kbA+7
    }
#pragma unroll
    for (int j = 0; j < 2; ++j) {
      const _Float16* wp = W + (size_t)(n0 + 16 * j + r) * K + k0 + kbB;
      b[j].h[0] = *(const v8h*)(wp);        // K = kbB .. kbB+7
      b[j].h[1] = *(const v8h*)(wp + 16);   // K = 16+kbB .. 16+kbB+7
    }
#pragma unroll
    for (int i = 0; i < 2; ++i)
#pragma unroll
      for (int j = 0; j < 2; ++j)
        acc[i][j] = WMMA_F32_F16(a[i].v, b[j].v, acc[i][j]);
  }
  // C/D layout: vgpr vi holds row m0+16i+vi+hi*8, lane r holds col n0+16j+r
#pragma unroll
  for (int i = 0; i < 2; ++i) {
#pragma unroll
    for (int j = 0; j < 2; ++j) {
      int n = n0 + 16 * j + r;
      float bv = bias ? bias[n] : 0.0f;
#pragma unroll
      for (int vi = 0; vi < 8; ++vi) {
        int m = m0 + 16 * i + vi + hi * 8;
        Y[(size_t)m * N + n] = acc[i][j][vi] + bv;
      }
    }
  }
}

// ---------------------------------------------------------------------------
// LayerNorm(q,k) + scale(q) + pack q,k -> f16 [B,H,N,Dh]; v -> f16 [B,H,Dh,N]
// One wave per (b,n,h); 64 elements -> 2 per lane; wave32 shuffle reduction.
// ---------------------------------------------------------------------------
__global__ __launch_bounds__(256) void lnpack_kernel(const float* __restrict__ qkv,
                                                     const float* __restrict__ g,
                                                     const float* __restrict__ bb,
                                                     _Float16* __restrict__ qh,
                                                     _Float16* __restrict__ kh,
                                                     _Float16* __restrict__ vt,
                                                     int Bn, int Nn, int Hn, float scale) {
  const int lane = threadIdx.x & 31;
  const int wave = threadIdx.x >> 5;
  int item = blockIdx.x * 8 + wave;          // over B*N*H
  int h = item % Hn;
  int t = item / Hn;
  int n = t % Nn;
  int b = t / Nn;
  const size_t rowbase = ((size_t)(b * Nn + n)) * 3072 + (size_t)h * 64;
  const int d = lane * 2;
  const float gd0 = g[d], gd1 = g[d + 1];
  const float bd0 = bb[d], bd1 = bb[d + 1];
  const size_t headbase = ((size_t)((b * Hn + h) * Nn + n)) * 64;

#pragma unroll
  for (int s = 0; s < 2; ++s) {  // s=0: q, s=1: k
    float2 xv = *(const float2*)(qkv + rowbase + (size_t)s * 1024 + d);
    float s1 = xv.x + xv.y;
    float s2 = xv.x * xv.x + xv.y * xv.y;
#pragma unroll
    for (int m = 16; m >= 1; m >>= 1) {
      s1 += __shfl_xor(s1, m);
      s2 += __shfl_xor(s2, m);
    }
    float mu = s1 * (1.0f / 64.0f);
    float var = s2 * (1.0f / 64.0f) - mu * mu;
    float rstd = rsqrtf(var + 1e-5f);
    float y0 = (xv.x - mu) * rstd * gd0 + bd0;
    float y1 = (xv.y - mu) * rstd * gd1 + bd1;
    if (s == 0) { y0 *= scale; y1 *= scale; }
    v2h hv = {(_Float16)y0, (_Float16)y1};
    _Float16* dst = (s == 0 ? qh : kh) + headbase + d;
    *(v2h*)dst = hv;
  }
  // v: plain convert, transposed layout [B,H,Dh,N]
  float2 vv = *(const float2*)(qkv + rowbase + 2048 + d);
  const size_t vtbase = ((size_t)(b * Hn + h) * 64);
  vt[(vtbase + d) * Nn + n]     = (_Float16)vv.x;
  vt[(vtbase + d + 1) * Nn + n] = (_Float16)vv.y;
}

// ---------------------------------------------------------------------------
// Flash attention. One wave handles 16 Q rows of one (b,h); 8 waves/block.
// Per 32-col KV block: S = Q*K^T (4 wmma), online softmax, O += P*V (4 wmma).
// ---------------------------------------------------------------------------
__global__ __launch_bounds__(256) void attn_kernel(const _Float16* __restrict__ qh,
                                                   const _Float16* __restrict__ kh,
                                                   const _Float16* __restrict__ vt,
                                                   _Float16* __restrict__ oh,
                                                   const int* __restrict__ causal_flag,
                                                   int Bn, int Hn, int Nn) {
  __shared__ alignas(16) _Float16 lds_p[8][16][40];  // 40-half row stride (pad)
  const int lane = threadIdx.x & 31;
  const int wave = threadIdx.x >> 5;
  const int nrb = Nn / 128;
  const int rowblk = blockIdx.x % nrb;
  const int bh = blockIdx.x / nrb;
  const int h = bh % Hn, b = bh / Hn;
  const int q0 = rowblk * 128 + wave * 16;
  const bool causal = (*causal_flag) != 0;
  const int r  = lane & 15;
  const int hi = lane >> 4;

  const size_t base = (size_t)bh * Nn * 64;
  const _Float16* Q  = qh + base;
  const _Float16* Kp = kh + base;
  const _Float16* Vt = vt + base;  // [64, Nn] for this (b,h)

  // Q A-fragments for Dh chunks [0,32) and [32,64)
  U16h qa[2];
  {
    const _Float16* qp = Q + (size_t)(q0 + r) * 64 + hi * 8;
    qa[0].h[0] = *(const v8h*)(qp);
    qa[0].h[1] = *(const v8h*)(qp + 16);
    qa[1].h[0] = *(const v8h*)(qp + 32);
    qa[1].h[1] = *(const v8h*)(qp + 48);
  }

  float rmax[8], rsum[8];
  v8f oacc[4] = {};
#pragma unroll
  for (int i = 0; i < 8; ++i) { rmax[i] = -3.0e38f; rsum[i] = 0.0f; }

  const int kend = causal ? (q0 + 16) : Nn;  // cols < kend
  for (int k0 = 0; k0 < kend; k0 += 32) {
    // ---- S = Q * K^T for two 16-col subtiles
    v8f S[2];
#pragma unroll
    for (int t = 0; t < 2; ++t) {
      const _Float16* kp = Kp + (size_t)(k0 + t * 16 + r) * 64 + hi * 8;
      U16h kf0, kf1;
      kf0.h[0] = *(const v8h*)(kp);
      kf0.h[1] = *(const v8h*)(kp + 16);
      kf1.h[0] = *(const v8h*)(kp + 32);
      kf1.h[1] = *(const v8h*)(kp + 48);
      v8f s = {};
      s = WMMA_F32_F16(qa[0].v, kf0.v, s);
      s = WMMA_F32_F16(qa[1].v, kf1.v, s);
      S[t] = s;
    }
    // ---- causal mask (element (m=i+hi*8, n=r) of each subtile)
    if (causal) {
#pragma unroll
      for (int t = 0; t < 2; ++t) {
        int col = k0 + t * 16 + r;
#pragma unroll
        for (int i = 0; i < 8; ++i) {
          int qr = q0 + i + hi * 8;
          if (col > qr) S[t][i] = -3.0e38f;
        }
      }
    }
    // ---- online softmax (row stats live redundantly in each 16-lane half)
    float corr[8];
#pragma unroll
    for (int i = 0; i < 8; ++i) {
      float mx = fmaxf(S[0][i], S[1][i]);
      mx = fmaxf(mx, __shfl_xor(mx, 1));
      mx = fmaxf(mx, __shfl_xor(mx, 2));
      mx = fmaxf(mx, __shfl_xor(mx, 4));
      mx = fmaxf(mx, __shfl_xor(mx, 8));
      float nm = fmaxf(rmax[i], mx);
      corr[i] = __expf(rmax[i] - nm);
      rmax[i] = nm;
      float p0 = __expf(S[0][i] - nm);
      float p1 = __expf(S[1][i] - nm);
      S[0][i] = p0; S[1][i] = p1;
      float ps = p0 + p1;
      ps += __shfl_xor(ps, 1);
      ps += __shfl_xor(ps, 2);
      ps += __shfl_xor(ps, 4);
      ps += __shfl_xor(ps, 8);
      rsum[i] = rsum[i] * corr[i] + ps;
    }
#pragma unroll
    for (int dt = 0; dt < 4; ++dt)
#pragma unroll
      for (int i = 0; i < 8; ++i) oacc[dt][i] *= corr[i];

    // ---- re-layout P (C-frag) -> A-frag through LDS (same-wave DS is in-order)
#pragma unroll
    for (int t = 0; t < 2; ++t)
#pragma unroll
      for (int i = 0; i < 8; ++i)
        lds_p[wave][i + hi * 8][t * 16 + r] = (_Float16)S[t][i];
    U16h pa;
    pa.h[0] = *(const v8h*)&lds_p[wave][r][hi * 8];
    pa.h[1] = *(const v8h*)&lds_p[wave][r][16 + hi * 8];

    // ---- O += P * V  (V^T gives contiguous B-fragments)
#pragma unroll
    for (int dt = 0; dt < 4; ++dt) {
      const _Float16* vp = Vt + (size_t)(dt * 16 + r) * Nn + k0 + hi * 8;
      U16h vf;
      vf.h[0] = *(const v8h*)(vp);
      vf.h[1] = *(const v8h*)(vp + 16);
      oacc[dt] = WMMA_F32_F16(pa.v, vf.v, oacc[dt]);
    }
  }

  // ---- finalize: divide by row sums, store O as f16 in [B*N, C] layout
#pragma unroll
  for (int i = 0; i < 8; ++i) rsum[i] = 1.0f / rsum[i];
#pragma unroll
  for (int dt = 0; dt < 4; ++dt) {
#pragma unroll
    for (int i = 0; i < 8; ++i) {
      int qr = q0 + i + hi * 8;
      float val = oacc[dt][i] * rsum[i];
      oh[((size_t)(b * Nn + qr)) * 1024 + h * 64 + dt * 16 + r] = (_Float16)val;
    }
  }
}

// ---------------------------------------------------------------------------
extern "C" void kernel_launch(void* const* d_in, const int* in_sizes, int n_in,
                              void* d_out, int out_size, void* d_ws, size_t ws_size,
                              hipStream_t stream) {
  const float* x     = (const float*)d_in[0];
  const float* Wqkv  = (const float*)d_in[1];
  const float* Wproj = (const float*)d_in[2];
  const float* bproj = (const float*)d_in[3];
  const float* lng   = (const float*)d_in[4];
  const float* lnb   = (const float*)d_in[5];
  const int*   causal = (const int*)d_in[6];
  float* out = (float*)d_out;

  constexpr int Bn = 2, Nn = 2048, C = 1024, Hn = 16;
  constexpr int MN = Bn * Nn;            // 4096 rows
  constexpr float SCALE = 8.0f / 64.0f;  // mup scale

  char* ws = (char*)d_ws;
  size_t off = 0;
  _Float16* xh     = (_Float16*)(ws + off); off += (size_t)MN * C * 2;
  _Float16* wqkvh  = (_Float16*)(ws + off); off += (size_t)3 * C * C * 2;
  _Float16* wprojh = (_Float16*)(ws + off); off += (size_t)C * C * 2;
  float*    qkv    = (float*)(ws + off);    off += (size_t)MN * 3 * C * 4;
  _Float16* qh     = (_Float16*)(ws + off); off += (size_t)MN * C * 2;
  _Float16* kh     = (_Float16*)(ws + off); off += (size_t)MN * C * 2;
  _Float16* vt     = (_Float16*)(ws + off); off += (size_t)MN * C * 2;
  _Float16* oh     = (_Float16*)(ws + off); off += (size_t)MN * C * 2;

  // 1) convert inputs to f16
  {
    int n4;
    n4 = MN * C / 4;
    cvt_f16_kernel<<<(n4 + 255) / 256, 256, 0, stream>>>(x, xh, n4);
    n4 = 3 * C * C / 4;
    cvt_f16_kernel<<<(n4 + 255) / 256, 256, 0, stream>>>(Wqkv, wqkvh, n4);
    n4 = C * C / 4;
    cvt_f16_kernel<<<(n4 + 255) / 256, 256, 0, stream>>>(Wproj, wprojh, n4);
  }
  // 2) QKV GEMM: [MN,C] x [3C,C]^T -> [MN,3C] fp32
  wmma_gemm_f16<<<dim3(MN / 64, 3 * C / 128), 256, 0, stream>>>(
      xh, wqkvh, qkv, nullptr, MN, 3 * C, C);
  // 3) LayerNorm + pack q/k/v
  lnpack_kernel<<<(Bn * Nn * Hn) / 8, 256, 0, stream>>>(
      qkv, lng, lnb, qh, kh, vt, Bn, Nn, Hn, SCALE);
  // 4) flash attention -> oh f16 [MN, C]
  attn_kernel<<<Bn * Hn * (Nn / 128), 256, 0, stream>>>(
      qh, kh, vt, oh, causal, Bn, Hn, Nn);
  // 5) projection GEMM with bias -> fp32 output
  wmma_gemm_f16<<<dim3(MN / 64, C / 128), 256, 0, stream>>>(
      oh, wprojh, out, bproj, MN, C, C);
}
